// Mlp_8744553415182
// MI455X (gfx1250) — compile-verified, matrix operand order checked by reading, not run on hardware
//
#include <hip/hip_runtime.h>

typedef float v2f __attribute__((ext_vector_type(2)));
typedef float v8f __attribute__((ext_vector_type(8)));
typedef int   v8i __attribute__((ext_vector_type(8)));
typedef int   v2i __attribute__((ext_vector_type(2)));
typedef int   v4i __attribute__((ext_vector_type(4)));
typedef unsigned int v4u __attribute__((ext_vector_type(4)));

// Problem constants (from reference setup_inputs)
constexpr int Dd   = 1024;                  // input dim
constexpr int Hh   = 4096;                  // hidden dim
constexpr int Mm   = 4 * 2048;              // B*S = 8192 rows
constexpr int TOPK = (int)(Dd * 0.1) + 1;   // 103
constexpr int THR  = (int)(Hh * 0.5);       // 2048 (strict >)

// LDS tile geometry. TDM pads 2 DWORDs after every 32 DWORDs stored, so each
// staged row occupies 34 DWORDs (136 B): 8B-aligned b64 fragment loads and
// conflict-free banks across the 16 fragment rows (34*r mod 64 all distinct).
constexpr int KCF   = 32;            // f32 K-chunk (elements) per stage
constexpr int KCQ   = 128;           // int8 K-chunk (bytes)    per stage
constexpr int ROWF  = 34;            // f32 LDS row stride, DWORDs
constexpr int ROWB  = 136;           // int8 LDS row stride, bytes
constexpr int ATILE = 32 * ROWB;     // 4352 B  (A: 32 rows)
constexpr int BTILE = 256 * ROWB;    // 34816 B (B: 256 rows)
constexpr int STAGE = ATILE + BTILE; // 39168 B per buffer (same for f32/int8)

// ---------------------------------------------------------------------------
// Tensor Data Mover: 2D tile -> LDS descriptor (ISA 08 §8.3/8.4).
// This toolchain exposes the 6-arg builtin:
//   (uint32x4 g0, int32x8 g1, int32x4 g2, int32x4 g3, int32x8, i32 cpol)
// One wave issues; completion via s_wait_tensorcnt.
//   data_size_code: 0=1B, 2=4B.  Stride/tile sizes in data_size units.
//   pad: interval 32 DWORDs (code 4), amount 2 DWORDs (code 1).
// ---------------------------------------------------------------------------
__device__ __forceinline__ void tdm_load_2d(unsigned lds_off, const void* gaddr,
                                            unsigned tile_k, unsigned tile_rows,
                                            unsigned long long row_stride_elems,
                                            unsigned data_size_code) {
  unsigned long long ga = (unsigned long long)(size_t)gaddr;
  v4u g0;
  g0.x = 1u;                                             // count=1, user mode
  g0.y = lds_off;                                        // lds_addr (bytes)
  g0.z = (unsigned)(ga & 0xffffffffu);                   // global_addr[31:0]
  g0.w = (unsigned)((ga >> 32) & 0x01ffffffu) | (2u << 30); // [56:32] | type=2
  v8i g1;
  g1[0] = (int)((data_size_code << 16) | (1u << 20) |    // data_size | pad_en
                (4u << 22) | (1u << 25));                // interval=32dw, pad=2dw
  g1[1] = (int)((tile_k & 0xffffu) << 16);               // tensor_dim0[15:0]
  g1[2] = (int)((tile_k >> 16) | ((tile_rows & 0xffffu) << 16)); // td0 hi|td1 lo
  g1[3] = (int)((tile_rows >> 16) | (tile_k << 16));     // td1 hi | tile_dim0
  g1[4] = (int)(tile_rows & 0xffffu);                    // tile_dim1 | tile_dim2=0
  g1[5] = (int)(row_stride_elems & 0xffffffffu);         // dim0_stride lo
  g1[6] = (int)((row_stride_elems >> 32) & 0xffffu);     // dim0_stride hi
  g1[7] = 0;
  v4i z4 = {0, 0, 0, 0};
  v8i z8 = {0, 0, 0, 0, 0, 0, 0, 0};
  __builtin_amdgcn_tensor_load_to_lds(g0, g1, z4, z4, z8, 0);
}

__device__ __forceinline__ unsigned ldso(const void* p) {
  return (unsigned)(size_t)p;   // LDS aperture: addr[31:0] is the LDS offset
}

// ---------------------------------------------------------------------------
// Fragment loaders (ISA 7.12.2 layouts, wave32)
// ---------------------------------------------------------------------------

// fp32 fragment from a padded LDS tile (row stride ROWF DWORDs).
__device__ __forceinline__ v2f ldsfrag_f32(const float* __restrict__ base,
                                           int row, int kc, int hi) {
  return *(const v2f*)(base + row * ROWF + kc + (hi << 1));
}

// int8 fragment (v8i) from a padded LDS tile (row stride ROWB bytes).
__device__ __forceinline__ v8i ldsfrag_i8(const char* __restrict__ base,
                                          int row, int kl, int hi) {
  const char* p = base + row * ROWB + kl + (hi << 3);
  v8i f;
#pragma unroll
  for (int j = 0; j < 4; ++j) {
    v2i d = *(const v2i*)(p + 16 * j);
    f[2 * j] = d.x;
    f[2 * j + 1] = d.y;
  }
  return f;
}

// Guarded global fp32 fragment (predictor K=103 tail, zero-fill past kmax).
__device__ __forceinline__ v2f ldfrag_f32_guard(const float* __restrict__ base, int ld,
                                                int row0, int kb, int lane, int kmax) {
  int r = row0 + (lane & 15);
  int k = kb + ((lane >> 4) << 1);
  const float* p = base + (size_t)r * ld + k;
  v2f out;
  out.x = (k     < kmax) ? p[0] : 0.0f;
  out.y = (k + 1 < kmax) ? p[1] : 0.0f;
  return out;
}

__device__ __forceinline__ float gelu_exact(float v) {
  return 0.5f * v * (1.0f + erff(v * 0.70710678118654752440f));
}

// ---------------------------------------------------------------------------
// Quantization kernels (per-row abs-max, RNE, clip [-128,127])
// ---------------------------------------------------------------------------

__device__ __forceinline__ int quant_clip(float v, float inv) {
  int q = __float2int_rn(v * inv);
  q = q > 127 ? 127 : q;
  q = q < -128 ? -128 : q;
  return q & 0xff;
}

__global__ void __launch_bounds__(256) quant_rows(const float* __restrict__ src,
                                                  signed char* __restrict__ qdst,
                                                  float* __restrict__ scales, int ncols) {
  int row = blockIdx.x;
  int t   = threadIdx.x;
  const float* rp = src + (size_t)row * ncols;
  float4 v = ((const float4*)rp)[t];
  float m = fmaxf(fmaxf(fabsf(v.x), fabsf(v.y)), fmaxf(fabsf(v.z), fabsf(v.w)));
#pragma unroll
  for (int o = 16; o > 0; o >>= 1) m = fmaxf(m, __shfl_xor(m, o, 32));
  __shared__ float red[8];
  if ((t & 31) == 0) red[t >> 5] = m;
  __syncthreads();
  if (t == 0) {
    float a = red[0];
#pragma unroll
    for (int i = 1; i < 8; ++i) a = fmaxf(a, red[i]);
    red[0] = fmaxf(a, 1e-5f) * (1.0f / 127.0f);
  }
  __syncthreads();
  float scale = red[0];
  float inv   = 1.0f / scale;
  int packed = quant_clip(v.x, inv) | (quant_clip(v.y, inv) << 8) |
               (quant_clip(v.z, inv) << 16) | (quant_clip(v.w, inv) << 24);
  ((int*)(qdst + (size_t)row * ncols))[t] = packed;
  if (t == 0) scales[row] = scale;
}

__global__ void __launch_bounds__(256) quant_bias(const float* __restrict__ b,
                                                  float* __restrict__ qb, int n) {
  int t = threadIdx.x;
  float m = 0.0f;
  for (int i = t; i < n; i += 256) m = fmaxf(m, fabsf(b[i]));
#pragma unroll
  for (int o = 16; o > 0; o >>= 1) m = fmaxf(m, __shfl_xor(m, o, 32));
  __shared__ float red[8];
  if ((t & 31) == 0) red[t >> 5] = m;
  __syncthreads();
  if (t == 0) {
    float a = red[0];
#pragma unroll
    for (int i = 1; i < 8; ++i) a = fmaxf(a, red[i]);
    red[0] = fmaxf(a, 1e-5f) * (1.0f / 127.0f);
  }
  __syncthreads();
  float scale = red[0];
  float inv   = 1.0f / scale;
  for (int i = t; i < n; i += 256) {
    int q = __float2int_rn(b[i] * inv);
    q = q > 127 ? 127 : q;
    q = q < -128 ? -128 : q;
    qb[i] = (float)q * scale;
  }
}

__global__ void zero_i32(int* __restrict__ p, int n) {
  int i = blockIdx.x * blockDim.x + threadIdx.x;
  if (i < n) p[i] = 0;
}

__global__ void select_mask(const int* __restrict__ np, int* __restrict__ mask,
                            int n, int thr) {
  int i = blockIdx.x * blockDim.x + threadIdx.x;
  if (i < n) mask[i] = (np[i] > thr) ? 1 : 0;
}

// ---------------------------------------------------------------------------
// Predictor: prod_topk = x[:, :103] @ W1[:, :103]^T + b1; count positives/ch.
// Small (3% of FLOPs): direct guarded loads, wave computes 16x64 strip.
// ---------------------------------------------------------------------------
__global__ void __launch_bounds__(256) predict_kernel(const float* __restrict__ x,
                                                      const float* __restrict__ W1,
                                                      const float* __restrict__ b1,
                                                      int* __restrict__ num_pos) {
  int lane   = threadIdx.x & 31;
  int gw     = blockIdx.x * 8 + (threadIdx.x >> 5);
  int tile_m = (gw & (Mm / 16 - 1)) * 16;
  int tile_n = (gw >> 9) * 64;

  v8f acc[4] = {};
  for (int kb = 0; kb < TOPK; kb += 4) {
    v2f a = ldfrag_f32_guard(x, Dd, tile_m, kb, lane, TOPK);
#pragma unroll
    for (int j = 0; j < 4; ++j) {
      v2f bf = ldfrag_f32_guard(W1, Dd, tile_n + 16 * j, kb, lane, TOPK);
      acc[j] = __builtin_amdgcn_wmma_f32_16x16x4_f32(false, a, false, bf,
                                                     (short)0, acc[j], false, false);
    }
  }
  int col = lane & 15;
#pragma unroll
  for (int j = 0; j < 4; ++j) {
    float bv = b1[tile_n + 16 * j + col];
    int cnt = 0;
#pragma unroll
    for (int r = 0; r < 8; ++r) cnt += ((acc[j][r] + bv) > 0.0f) ? 1 : 0;
    cnt += __shfl_xor(cnt, 16, 32);
    if (lane < 16) atomicAdd(&num_pos[tile_n + 16 * j + lane], cnt);
  }
}

// ---------------------------------------------------------------------------
// GEMM1: block = 8 waves, C tile 32(M) x 256(N); per-block path selection.
// fp path: TDM double-buffered fp32 tiles -> V_WMMA_F32_16X16X4_F32.
// q  path: TDM double-buffered int8 tiles -> V_WMMA_I32_16X16X64_IU8 (exact),
//          rescaled by sx[m]*sw[n] + qb[n]. Exact GELU; h -> workspace.
// ---------------------------------------------------------------------------
__global__ void __launch_bounds__(256) mlp1_kernel(
    const float* __restrict__ x, const float* __restrict__ W1,
    const float* __restrict__ b1, const signed char* __restrict__ qx,
    const float* __restrict__ sx, const signed char* __restrict__ qw,
    const float* __restrict__ swl, const float* __restrict__ qb,
    const int* __restrict__ fp_mask, float* __restrict__ hbuf) {
  __shared__ __align__(16) char smem[2][STAGE];
  __shared__ int s_need[2];

  int lane = threadIdx.x & 31;
  int wave = threadIdx.x >> 5;
  int bm   = (blockIdx.x & (Mm / 32 - 1)) * 32;   // 256 M-blocks
  int bn   = (blockIdx.x >> 8) * 256;             // 16 N-blocks
  int wm   = (wave & 1) * 16;
  int wn   = (wave >> 1) * 64;
  int col = lane & 15, hi = lane >> 4;

  // Per-column masks for this wave's 64 columns + block-level need flags.
  if (threadIdx.x == 0) { s_need[0] = 0; s_need[1] = 0; }
  __syncthreads();
  int maskv[4];
  int wfp = 0, wq = 0;
#pragma unroll
  for (int j = 0; j < 4; ++j) {
    maskv[j] = fp_mask[bn + wn + 16 * j + col];
    wfp |= __any(maskv[j] != 0);
    wq  |= __any(maskv[j] == 0);
  }
  if (lane == 0) {
    if (wfp) atomicOr(&s_need[0], 1);
    if (wq)  atomicOr(&s_need[1], 1);
  }
  __syncthreads();
  int need_fp = s_need[0], need_q = s_need[1];

  v8f accF[4] = {};
  v8i accI[4] = {};

  if (need_fp) {
    constexpr int NT = Dd / KCF;   // 32 stages
    if (wave == 0) {
      tdm_load_2d(ldso(&smem[0][0]),     x  + (size_t)bm * Dd, KCF, 32,  Dd, 2);
      tdm_load_2d(ldso(&smem[0][ATILE]), W1 + (size_t)bn * Dd, KCF, 256, Dd, 2);
      __builtin_amdgcn_s_wait_tensorcnt(0);
    }
    __syncthreads();
    for (int t = 0; t < NT; ++t) {
      int cur = t & 1, nxt = cur ^ 1;
      if (wave == 0 && t + 1 < NT) {
        int kb = (t + 1) * KCF;
        tdm_load_2d(ldso(&smem[nxt][0]),     x  + (size_t)bm * Dd + kb, KCF, 32,  Dd, 2);
        tdm_load_2d(ldso(&smem[nxt][ATILE]), W1 + (size_t)bn * Dd + kb, KCF, 256, Dd, 2);
      }
      const float* A = (const float*)&smem[cur][0];
      const float* B = (const float*)&smem[cur][ATILE];
#pragma unroll
      for (int kc = 0; kc < KCF; kc += 4) {
        v2f a = ldsfrag_f32(A, wm + col, kc, hi);
#pragma unroll
        for (int j = 0; j < 4; ++j) {
          v2f bf = ldsfrag_f32(B, wn + 16 * j + col, kc, hi);
          accF[j] = __builtin_amdgcn_wmma_f32_16x16x4_f32(false, a, false, bf,
                                                          (short)0, accF[j], false, false);
        }
      }
      if (wave == 0 && t + 1 < NT) __builtin_amdgcn_s_wait_tensorcnt(0);
      __syncthreads();
    }
  }

  if (need_q) {
    constexpr int NT = Dd / KCQ;   // 8 stages
    if (wave == 0) {
      tdm_load_2d(ldso(&smem[0][0]),     qx + (size_t)bm * Dd, KCQ, 32,  Dd, 0);
      tdm_load_2d(ldso(&smem[0][ATILE]), qw + (size_t)bn * Dd, KCQ, 256, Dd, 0);
      __builtin_amdgcn_s_wait_tensorcnt(0);
    }
    __syncthreads();
    for (int t = 0; t < NT; ++t) {
      int cur = t & 1, nxt = cur ^ 1;
      if (wave == 0 && t + 1 < NT) {
        int kb = (t + 1) * KCQ;
        tdm_load_2d(ldso(&smem[nxt][0]),     qx + (size_t)bm * Dd + kb, KCQ, 32,  Dd, 0);
        tdm_load_2d(ldso(&smem[nxt][ATILE]), qw + (size_t)bn * Dd + kb, KCQ, 256, Dd, 0);
      }
      const char* A8 = (const char*)&smem[cur][0];
      const char* B8 = (const char*)&smem[cur][ATILE];
#pragma unroll
      for (int kl = 0; kl < KCQ; kl += 64) {
        v8i a = ldsfrag_i8(A8, wm + col, kl, hi);
#pragma unroll
        for (int j = 0; j < 4; ++j) {
          v8i bf = ldsfrag_i8(B8, wn + 16 * j + col, kl, hi);
          accI[j] = __builtin_amdgcn_wmma_i32_16x16x64_iu8(true, a, true, bf,
                                                           accI[j], false, false);
        }
      }
      if (wave == 0 && t + 1 < NT) __builtin_amdgcn_s_wait_tensorcnt(0);
      __syncthreads();
    }
  }

#pragma unroll
  for (int j = 0; j < 4; ++j) {
    int   n   = bn + wn + 16 * j + col;
    float bv  = b1[n];
    float swn = swl[n];
    float qbn = qb[n];
    int   msk = maskv[j];
#pragma unroll
    for (int r = 0; r < 8; ++r) {
      int m = bm + wm + r + hi * 8;
      float val = msk ? (accF[j][r] + bv)
                      : ((float)accI[j][r] * sx[m] * swn + qbn);
      hbuf[(size_t)m * Hh + n] = gelu_exact(val);
    }
  }
}

// ---------------------------------------------------------------------------
// GEMM2: out = h @ W2^T + b2. Same TDM double-buffered pipeline, K = 4096.
// h (128 MB) stays resident in the 192 MB L2 between GEMM1 and GEMM2.
// ---------------------------------------------------------------------------
__global__ void __launch_bounds__(256) mlp2_kernel(const float* __restrict__ hbuf,
                                                   const float* __restrict__ W2,
                                                   const float* __restrict__ b2,
                                                   float* __restrict__ out) {
  __shared__ __align__(16) char smem[2][STAGE];
  int lane = threadIdx.x & 31;
  int wave = threadIdx.x >> 5;
  int bm   = (blockIdx.x & (Mm / 32 - 1)) * 32;
  int bn   = (blockIdx.x >> 8) * 256;             // 4 N-blocks over D=1024
  int wm   = (wave & 1) * 16;
  int wn   = (wave >> 1) * 64;
  int col = lane & 15, hi = lane >> 4;

  constexpr int NT = Hh / KCF;   // 128 stages
  if (wave == 0) {
    tdm_load_2d(ldso(&smem[0][0]),     hbuf + (size_t)bm * Hh, KCF, 32,  Hh, 2);
    tdm_load_2d(ldso(&smem[0][ATILE]), W2   + (size_t)bn * Hh, KCF, 256, Hh, 2);
    __builtin_amdgcn_s_wait_tensorcnt(0);
  }
  __syncthreads();

  v8f acc[4] = {};
  for (int t = 0; t < NT; ++t) {
    int cur = t & 1, nxt = cur ^ 1;
    if (wave == 0 && t + 1 < NT) {
      int kb = (t + 1) * KCF;
      tdm_load_2d(ldso(&smem[nxt][0]),     hbuf + (size_t)bm * Hh + kb, KCF, 32,  Hh, 2);
      tdm_load_2d(ldso(&smem[nxt][ATILE]), W2   + (size_t)bn * Hh + kb, KCF, 256, Hh, 2);
    }
    const float* A = (const float*)&smem[cur][0];
    const float* B = (const float*)&smem[cur][ATILE];
#pragma unroll
    for (int kc = 0; kc < KCF; kc += 4) {
      v2f a = ldsfrag_f32(A, wm + col, kc, hi);
#pragma unroll
      for (int j = 0; j < 4; ++j) {
        v2f bf = ldsfrag_f32(B, wn + 16 * j + col, kc, hi);
        acc[j] = __builtin_amdgcn_wmma_f32_16x16x4_f32(false, a, false, bf,
                                                       (short)0, acc[j], false, false);
      }
    }
    if (wave == 0 && t + 1 < NT) __builtin_amdgcn_s_wait_tensorcnt(0);
    __syncthreads();
  }

#pragma unroll
  for (int j = 0; j < 4; ++j) {
    int n = bn + wn + 16 * j + col;
    float bv = b2[n];
#pragma unroll
    for (int r = 0; r < 8; ++r) {
      int m = bm + wm + r + hi * 8;
      out[(size_t)m * Dd + n] = acc[j][r] + bv;
    }
  }
}

// ---------------------------------------------------------------------------
// Host launcher
// ---------------------------------------------------------------------------
extern "C" void kernel_launch(void* const* d_in, const int* in_sizes, int n_in,
                              void* d_out, int out_size, void* d_ws, size_t ws_size,
                              hipStream_t stream) {
  const float* x  = (const float*)d_in[0];
  const float* W1 = (const float*)d_in[1];
  const float* b1 = (const float*)d_in[2];
  const float* W2 = (const float*)d_in[3];
  const float* b2 = (const float*)d_in[4];
  float* out = (float*)d_out;

  char*  ws  = (char*)d_ws;
  size_t off = 0;
  auto walloc = [&](size_t bytes) -> void* {
    void* p = ws + off;
    off = (off + bytes + 255) & ~(size_t)255;
    return p;
  };
  signed char* qx      = (signed char*)walloc((size_t)Mm * Dd);
  float*       sx      = (float*)walloc((size_t)Mm * 4);
  signed char* qw      = (signed char*)walloc((size_t)Hh * Dd);
  float*       swl     = (float*)walloc((size_t)Hh * 4);
  float*       qb      = (float*)walloc((size_t)Hh * 4);
  int*         num_pos = (int*)walloc((size_t)Hh * 4);
  int*         fpmask  = (int*)walloc((size_t)Hh * 4);
  float*       hbuf    = (float*)walloc((size_t)Mm * Hh * 4);
  (void)ws_size; (void)in_sizes; (void)n_in; (void)out_size;

  quant_rows<<<Mm, 256, 0, stream>>>(x, qx, sx, Dd);
  quant_rows<<<Hh, 256, 0, stream>>>(W1, qw, swl, Dd);
  quant_bias<<<1, 256, 0, stream>>>(b1, qb, Hh);
  zero_i32<<<(Hh + 255) / 256, 256, 0, stream>>>(num_pos, Hh);

  predict_kernel<<<(Mm / 16) * (Hh / 64) / 8, 256, 0, stream>>>(x, W1, b1, num_pos);
  select_mask<<<(Hh + 255) / 256, 256, 0, stream>>>(num_pos, fpmask, Hh, THR);

  mlp1_kernel<<<(Mm / 32) * (Hh / 256), 256, 0, stream>>>(x, W1, b1, qx, sx, qw, swl,
                                                          qb, fpmask, hbuf);
  mlp2_kernel<<<(Mm / 32) * (Dd / 256), 256, 0, stream>>>(hbuf, W2, b2, out);
}
